// SequentialDecoder_11295763988763
// MI455X (gfx1250) — matrix operand order repaired and checked by reading, not compiled, not run on hardware
//
#include <hip/hip_runtime.h>
#include <hip/hip_bf16.h>
#include <cstdint>

// ---------- types for WMMA (gfx1250, wave32) ----------
typedef __attribute__((ext_vector_type(16))) __bf16 v16bf;
typedef __attribute__((ext_vector_type(8)))  __bf16 v8bf;
typedef __attribute__((ext_vector_type(8)))  float  v8f;

#define BATCH    512
#define D_MODEL  1024
#define C_OUT    64
#define PRED_LEN 96
#define D_INNER  2048
#define DT_RANK  64
#define D_STATE  64
#define N_XZ     4096   // 2*D_INNER
#define N_XDBC   192    // DT_RANK + 2*D_STATE

__device__ __forceinline__ float silu_f(float v)     { return v / (1.0f + __expf(-v)); }
__device__ __forceinline__ float softplus_f(float v) { return v > 20.0f ? v : log1pf(__expf(v)); }

__device__ __forceinline__ v8f wmma_bf(v16bf a, v16bf b, v8f c) {
  // D = A(16x32 bf16) * B(32x16 bf16) + C(16x16 f32)
  return __builtin_amdgcn_wmma_f32_16x16x32_bf16(false, a, false, b, (short)0, c, false, false);
}

// A fragment from row-major bf16 [M x lda] (global or LDS):
// lane m=lane%16, khalf=lane/16
// a[0..15] = A[m][k0 + 16*khalf + 0..15]
__device__ __forceinline__ v16bf load_a_frag(const __bf16* A, int lda, int row0, int k0) {
  int lane = threadIdx.x & 31;
  const __bf16* p = A + (size_t)(row0 + (lane & 15)) * lda + k0 + ((lane >> 4) << 4);
  v8bf lo = *(const v8bf*)(p);
  v8bf hi = *(const v8bf*)(p + 8);
  v16bf a;
#pragma unroll
  for (int i = 0; i < 8; ++i) { a[i] = lo[i]; a[8 + i] = hi[i]; }
  return a;
}

// B fragment from K-panel-packed weights Wp[(k/16)][n][16] (bf16):
// lane n=lane%16, khalf=lane/16, b[e] = W[k0 + 16*khalf + e][n] -> 32B contiguous
__device__ __forceinline__ v16bf load_b_frag(const __bf16* Bp, int N, int k0, int col0) {
  int lane = threadIdx.x & 31;
  const __bf16* p = Bp + ((size_t)((k0 >> 4) + (lane >> 4)) * N + (col0 + (lane & 15))) * 16;
  v8bf lo = *(const v8bf*)(p);
  v8bf hi = *(const v8bf*)(p + 8);
  v16bf b;
#pragma unroll
  for (int i = 0; i < 8; ++i) { b[i] = lo[i]; b[8 + i] = hi[i]; }
  return b;
}

// ---------- preprocessing ----------
__global__ void cvt_bf16_kernel(const float* __restrict__ src, __bf16* __restrict__ dst, int n) {
  int i = blockIdx.x * 256 + threadIdx.x;
  if (i < n) dst[i] = (__bf16)src[i];
}

// W row-major [K x N] f32  ->  Wp[(k/16)*N*16 + n*16 + (k%16)] bf16
__global__ void pack_b_kernel(const float* __restrict__ W, __bf16* __restrict__ Wp, int K, int N) {
  int i = blockIdx.x * 256 + threadIdx.x;
  if (i >= K * N) return;
  int k = i / N, n = i - k * N;
  Wp[((size_t)(k >> 4) * N + n) * 16 + (k & 15)] = (__bf16)W[i];
}

// ctx_part = context_bf16 @ w_in_head + b_in : [512,1024]x[1024,4096] -> f32
__global__ void gemm_ctx_kernel(const __bf16* __restrict__ A, const __bf16* __restrict__ Bp,
                                const float* __restrict__ bias, float* __restrict__ C) {
  int wave = threadIdx.x >> 5;
  int row0 = blockIdx.y * 16;
  int col0 = blockIdx.x * 256 + wave * 64;
  v8f acc[4] = {};
  for (int k0 = 0; k0 < D_MODEL; k0 += 32) {
    v16bf a = load_a_frag(A, D_MODEL, row0, k0);
#pragma unroll
    for (int t = 0; t < 4; ++t)
      acc[t] = wmma_bf(a, load_b_frag(Bp, N_XZ, k0, col0 + t * 16), acc[t]);
  }
  int lane = threadIdx.x & 31;
  int r0 = row0 + ((lane >> 4) << 3);
  int c0 = col0 + (lane & 15);
#pragma unroll
  for (int t = 0; t < 4; ++t) {
    int c = c0 + t * 16;
    float bv = bias[c];
#pragma unroll
    for (int r = 0; r < 8; ++r)
      C[(size_t)(r0 + r) * N_XZ + c] = acc[t][r] + bv;
  }
}

// K1: xz = ctx_part + pred @ w_in_tail; x=silu(conv), z=silu(z)
__global__ void k1_xz_kernel(const __bf16* __restrict__ PredBf, const __bf16* __restrict__ WtP,
                             const float* __restrict__ ctx_part,
                             const float* __restrict__ conv_w, const float* __restrict__ conv_b,
                             float* __restrict__ x_f32, __bf16* __restrict__ x_bf,
                             float* __restrict__ z_f32) {
  int wave = threadIdx.x >> 5;
  int row0 = blockIdx.y * 16;
  int col0 = blockIdx.x * 256 + wave * 64;
  v8f acc[4] = {};
#pragma unroll
  for (int k0 = 0; k0 < C_OUT; k0 += 32) {
    v16bf a = load_a_frag(PredBf, C_OUT, row0, k0);
#pragma unroll
    for (int t = 0; t < 4; ++t)
      acc[t] = wmma_bf(a, load_b_frag(WtP, N_XZ, k0, col0 + t * 16), acc[t]);
  }
  int lane = threadIdx.x & 31;
  int r0 = row0 + ((lane >> 4) << 3);
  int c0 = col0 + (lane & 15);
#pragma unroll
  for (int t = 0; t < 4; ++t) {
    int c = c0 + t * 16;
#pragma unroll
    for (int r = 0; r < 8; ++r) {
      int row = r0 + r;
      float v = acc[t][r] + ctx_part[(size_t)row * N_XZ + c];
      if (c < D_INNER) {
        float xs = v * conv_w[c * 4 + 3] + conv_b[c];
        float x = silu_f(xs);
        x_f32[(size_t)row * D_INNER + c] = x;
        x_bf [(size_t)row * D_INNER + c] = (__bf16)x;
      } else {
        z_f32[(size_t)row * D_INNER + (c - D_INNER)] = silu_f(v);
      }
    }
  }
}

// K23 (fused): xdbc = x @ w_x (16-row tile) -> LDS; s = sum(Bm*Cm); dt tile -> bf16 LDS;
//              dt = softplus(dt_tile @ w_dt + b_dt); y = (dt*x*s + D_skip*x)*silu(z) -> y_bf
__global__ void k23_kernel(const __bf16* __restrict__ Xbf, const __bf16* __restrict__ WxP,
                           const __bf16* __restrict__ WdtP, const float* __restrict__ b_dt,
                           const float* __restrict__ D_skip, const float* __restrict__ x_f32,
                           const float* __restrict__ z_f32, __bf16* __restrict__ y_bf) {
  __shared__ float  xdbc[16][200];   // padded (stride 800B -> bank shift 8)
  __shared__ __bf16 dtb[16][80];     // padded (stride 160B, 16B aligned)
  __shared__ float  s_sh[16];
  int wave = threadIdx.x >> 5;
  int row0 = blockIdx.x * 16;
  int lane = threadIdx.x & 31;
  int r0l  = (lane >> 4) << 3;
  int cb   = lane & 15;

  // phase 1: xdbc tile [16 x 192]
  {
    v8f acc[3] = {};
    for (int k0 = 0; k0 < D_INNER; k0 += 32) {
      v16bf a = load_a_frag(Xbf, D_INNER, row0, k0);
#pragma unroll
      for (int t = 0; t < 3; ++t)
        acc[t] = wmma_bf(a, load_b_frag(WxP, N_XDBC, k0, wave * 48 + t * 16), acc[t]);
    }
#pragma unroll
    for (int t = 0; t < 3; ++t)
#pragma unroll
      for (int r = 0; r < 8; ++r)
        xdbc[r0l + r][wave * 48 + t * 16 + cb] = acc[t][r];
  }
  __syncthreads();

  // phase 2: s[row] and bf16 dt tile
  for (int i = threadIdx.x; i < 16 * DT_RANK; i += 128) {
    int r = i >> 6, c = i & 63;
    dtb[r][c] = (__bf16)xdbc[r][c];
  }
  if (threadIdx.x < 16) {
    int r = threadIdx.x;
    float s = 0.f;
#pragma unroll 4
    for (int i = 0; i < D_STATE; ++i)
      s += xdbc[r][DT_RANK + i] * xdbc[r][DT_RANK + D_STATE + i];
    s_sh[r] = s;
  }
  __syncthreads();

  // phase 3: dt GEMM (A from LDS, K=64) + y epilogue; per wave 512 cols in 8 chunks
  for (int chunk = 0; chunk < 8; ++chunk) {
    int col0 = wave * 512 + chunk * 64;
    v8f acc[4] = {};
#pragma unroll
    for (int k0 = 0; k0 < DT_RANK; k0 += 32) {
      v16bf a = load_a_frag(&dtb[0][0], 80, 0, k0);
#pragma unroll
      for (int t = 0; t < 4; ++t)
        acc[t] = wmma_bf(a, load_b_frag(WdtP, D_INNER, k0, col0 + t * 16), acc[t]);
    }
#pragma unroll
    for (int t = 0; t < 4; ++t) {
      int c = col0 + t * 16 + cb;
      float bd = b_dt[c], dk = D_skip[c];
#pragma unroll
      for (int r = 0; r < 8; ++r) {
        int rl = r0l + r;
        int row = row0 + rl;
        float dt = softplus_f(acc[t][r] + bd);
        float x = x_f32[(size_t)row * D_INNER + c];
        float y = (dt * x * s_sh[rl] + dk * x) * z_f32[(size_t)row * D_INNER + c];
        y_bf[(size_t)row * D_INNER + c] = (__bf16)y;
      }
    }
  }
}

// K45 (fused): h tile = y @ w_out + b_out (16 rows x 1024 cols) -> LDS (bf16);
//              pred = h_tile @ w_proj + b_proj -> out[:,t,:] + pred_bf
__global__ void k45_kernel(const __bf16* __restrict__ Ybf, const __bf16* __restrict__ WoP,
                           const float* __restrict__ b_out, const __bf16* __restrict__ WpP,
                           const float* __restrict__ b_proj, float* __restrict__ out,
                           __bf16* __restrict__ pred_bf, int t) {
  __shared__ __bf16 hsh[16][D_MODEL + 16];   // stride 2080B: 16B aligned, bank shift 8/row
  int wave = threadIdx.x >> 5;
  int row0 = blockIdx.x * 16;
  int lane = threadIdx.x & 31;
  int r0l  = (lane >> 4) << 3;
  int cb   = lane & 15;

  // phase 1: h tile; each wave 256 cols = 4 chunks of 64
  for (int chunk = 0; chunk < 4; ++chunk) {
    int col0 = wave * 256 + chunk * 64;
    v8f acc[4] = {};
    for (int k0 = 0; k0 < D_INNER; k0 += 32) {
      v16bf a = load_a_frag(Ybf, D_INNER, row0, k0);
#pragma unroll
      for (int tt = 0; tt < 4; ++tt)
        acc[tt] = wmma_bf(a, load_b_frag(WoP, D_MODEL, k0, col0 + tt * 16), acc[tt]);
    }
#pragma unroll
    for (int tt = 0; tt < 4; ++tt) {
      int c = col0 + tt * 16 + cb;
      float bv = b_out[c];
#pragma unroll
      for (int r = 0; r < 8; ++r)
        hsh[r0l + r][c] = (__bf16)(acc[tt][r] + bv);
    }
  }
  __syncthreads();

  // phase 2: pred = h_tile @ w_proj (A fragments from LDS, K=1024); one 16-col tile per wave
  {
    int col0 = wave * 16;
    v8f acc = {};
    for (int k0 = 0; k0 < D_MODEL; k0 += 32)
      acc = wmma_bf(load_a_frag(&hsh[0][0], D_MODEL + 16, 0, k0),
                    load_b_frag(WpP, C_OUT, k0, col0), acc);
    int c = col0 + cb;
    float bv = b_proj[c];
#pragma unroll
    for (int r = 0; r < 8; ++r) {
      int row = row0 + r0l + r;
      float v = acc[r] + bv;
      out[(size_t)row * (PRED_LEN * C_OUT) + t * C_OUT + c] = v;
      pred_bf[(size_t)row * C_OUT + c] = (__bf16)v;
    }
  }
}

// ---------- host ----------
extern "C" void kernel_launch(void* const* d_in, const int* in_sizes, int n_in,
                              void* d_out, int out_size, void* d_ws, size_t ws_size,
                              hipStream_t stream) {
  const float* context        = (const float*)d_in[0];   // [512,1024]
  const float* initial_values = (const float*)d_in[1];   // [512,64]
  const float* w_in           = (const float*)d_in[2];   // [1088,4096]
  const float* b_in           = (const float*)d_in[3];   // [4096]
  const float* conv_w         = (const float*)d_in[4];   // [2048,4]
  const float* conv_b         = (const float*)d_in[5];   // [2048]
  const float* w_x            = (const float*)d_in[6];   // [2048,192]
  const float* w_dt           = (const float*)d_in[7];   // [64,2048]
  const float* b_dt           = (const float*)d_in[8];   // [2048]
  /* d_in[9] = A_log, unused (L=1, h0=0) */
  const float* D_skip         = (const float*)d_in[10];  // [2048]
  const float* w_out          = (const float*)d_in[11];  // [2048,1024]
  const float* b_out          = (const float*)d_in[12];  // [1024]
  const float* w_proj         = (const float*)d_in[13];  // [1024,64]
  const float* b_proj         = (const float*)d_in[14];  // [64]
  float* out = (float*)d_out;

  // workspace carve-out (256B aligned slices)
  size_t off = 0;
  char* base = (char*)d_ws;
  auto alloc = [&](size_t bytes) -> char* {
    char* p = base + off;
    off += (bytes + 255) & ~(size_t)255;
    return p;
  };
  float*  ctx_part = (float*) alloc((size_t)BATCH * N_XZ * 4);
  __bf16* ctxA     = (__bf16*)alloc((size_t)BATCH * D_MODEL * 2);
  __bf16* winHeadP = (__bf16*)alloc((size_t)D_MODEL * N_XZ * 2);
  __bf16* winTailP = (__bf16*)alloc((size_t)C_OUT * N_XZ * 2);
  __bf16* wxP      = (__bf16*)alloc((size_t)D_INNER * N_XDBC * 2);
  __bf16* wdtP     = (__bf16*)alloc((size_t)DT_RANK * D_INNER * 2);
  __bf16* woutP    = (__bf16*)alloc((size_t)D_INNER * D_MODEL * 2);
  __bf16* wprojP   = (__bf16*)alloc((size_t)D_MODEL * C_OUT * 2);
  float*  x_f32    = (float*) alloc((size_t)BATCH * D_INNER * 4);
  __bf16* x_bf     = (__bf16*)alloc((size_t)BATCH * D_INNER * 2);
  float*  z_f32    = (float*) alloc((size_t)BATCH * D_INNER * 4);
  __bf16* y_bf     = (__bf16*)alloc((size_t)BATCH * D_INNER * 2);
  __bf16* pred_bf  = (__bf16*)alloc((size_t)BATCH * C_OUT * 2);

  auto grid1 = [](size_t n) { return dim3((unsigned)((n + 255) / 256)); };

  // --- preprocessing (deterministic each call) ---
  pack_b_kernel<<<grid1((size_t)D_MODEL * N_XZ), 256, 0, stream>>>(w_in, winHeadP, D_MODEL, N_XZ);
  pack_b_kernel<<<grid1((size_t)C_OUT * N_XZ), 256, 0, stream>>>(w_in + (size_t)D_MODEL * N_XZ, winTailP, C_OUT, N_XZ);
  pack_b_kernel<<<grid1((size_t)D_INNER * N_XDBC), 256, 0, stream>>>(w_x, wxP, D_INNER, N_XDBC);
  pack_b_kernel<<<grid1((size_t)DT_RANK * D_INNER), 256, 0, stream>>>(w_dt, wdtP, DT_RANK, D_INNER);
  pack_b_kernel<<<grid1((size_t)D_INNER * D_MODEL), 256, 0, stream>>>(w_out, woutP, D_INNER, D_MODEL);
  pack_b_kernel<<<grid1((size_t)D_MODEL * C_OUT), 256, 0, stream>>>(w_proj, wprojP, D_MODEL, C_OUT);
  cvt_bf16_kernel<<<grid1((size_t)BATCH * D_MODEL), 256, 0, stream>>>(context, ctxA, BATCH * D_MODEL);
  cvt_bf16_kernel<<<grid1((size_t)BATCH * C_OUT), 256, 0, stream>>>(initial_values, pred_bf, BATCH * C_OUT);

  // hoisted: ctx_part = context @ w_in[:1024] + b_in
  gemm_ctx_kernel<<<dim3(N_XZ / 256, BATCH / 16), 128, 0, stream>>>(ctxA, winHeadP, b_in, ctx_part);

  // --- 96 sequential decode steps: 3 dependent kernels per step ---
  for (int t = 0; t < PRED_LEN; ++t) {
    k1_xz_kernel<<<dim3(N_XZ / 256, BATCH / 16), 128, 0, stream>>>(
        pred_bf, winTailP, ctx_part, conv_w, conv_b, x_f32, x_bf, z_f32);
    k23_kernel<<<dim3(BATCH / 16), 128, 0, stream>>>(
        x_bf, wxP, wdtP, b_dt, D_skip, x_f32, z_f32, y_bf);
    k45_kernel<<<dim3(BATCH / 16), 128, 0, stream>>>(
        y_bf, woutP, b_out, wprojP, b_proj, out, pred_bf, t);
  }
}
